// RQVAE_40089224740884
// MI455X (gfx1250) — matrix operand order repaired and checked by reading, not compiled, not run on hardware
//
#include <hip/hip_runtime.h>
#include <hip/hip_bf16.h>

// ---------------------------------------------------------------------------
// RQ-VAE forward for MI455X (gfx1250, wave32, WMMA).
// Encoder MLP -> RVQ (8 codebooks x 256 x 32) -> Decoder MLP.
// GEMMs use v_wmma_f32_16x16x32_f16 (f16 inputs, f32 accumulate).
// Wave tile = 32 rows x NT*16 cols (2 M-subtiles reuse each B fragment).
// ---------------------------------------------------------------------------

typedef __attribute__((ext_vector_type(16))) _Float16 v16h;
typedef __attribute__((ext_vector_type(8)))  _Float16 v8h;
typedef __attribute__((ext_vector_type(8)))  float    v8f;

#define BATCH_N   131072
#define NUM_CB    8
#define CB_SIZE   256
#define LAT_D     32

// ---------------------------------------------------------------------------
// Weight transpose + f32 -> f16 convert: WT[n*K + k] = (f16) W[k*N + n]
// ---------------------------------------------------------------------------
__global__ __launch_bounds__(256) void convert_wt_kernel(
    const float* __restrict__ W, _Float16* __restrict__ WT, int K, int N) {
  int i = blockIdx.x * 256 + threadIdx.x;
  if (i < K * N) {
    int n = i / K;
    int k = i - n * K;
    WT[(size_t)n * K + k] = (_Float16)W[(size_t)k * N + n];
  }
}

// ---------------------------------------------------------------------------
// WMMA GEMM:  out = act(A[M,K] @ W[K,N] + bias)
//   A: row-major, element type TA (float or _Float16), converted to f16 frags
//   WT: f16, transposed [N,K] row-major (B-fragment loads are contiguous)
//   block = 256 threads (8 waves)
//   wave tile  = 32 rows x NT*16 cols (MT=2 M-subtiles share each B frag)
//   block tile = 256 rows x NT*16 cols; grid = (N/(NT*16), M/256)
//   N multiple of NT*16; K multiple of 32; M multiple of 256. Branch-free loop.
// ---------------------------------------------------------------------------
template <typename TA, bool RELU, int NT>
__global__ __launch_bounds__(256) void gemm_wmma_kernel(
    const TA* __restrict__ A, const _Float16* __restrict__ WT,
    const float* __restrict__ bias, int M, int N, int K,
    _Float16* __restrict__ out16, float* __restrict__ out32) {
  constexpr int MT = 2;
  const int lane = threadIdx.x & 31;
  const int wave = threadIdx.x >> 5;
  const int half = (lane >> 4) & 1;   // 0: lanes 0-15, 1: lanes 16-31
  const int l15  = lane & 15;
  const int m0   = blockIdx.y * 256 + wave * 32;
  const int n0   = blockIdx.x * (NT * 16);

  v8f acc[MT][NT];
#pragma unroll
  for (int mt = 0; mt < MT; ++mt) {
#pragma unroll
    for (int t = 0; t < NT; ++t) {
#pragma unroll
      for (int r = 0; r < 8; ++r) acc[mt][t][r] = 0.0f;
    }
  }

  // Per-lane base pointers (hoisted out of the k-loop).
  const TA* ap[MT];
#pragma unroll
  for (int mt = 0; mt < MT; ++mt)
    ap[mt] = A + (size_t)(m0 + mt * 16 + l15) * K + half * 16;
  const _Float16* bp[NT];
#pragma unroll
  for (int t = 0; t < NT; ++t)
    bp[t] = WT + (size_t)(n0 + t * 16 + l15) * K + half * 16;

#pragma unroll 2
  for (int k0 = 0; k0 < K; k0 += 32) {
    // ---- MT A fragments (contiguous k = 16*half + s, same order as B) ----
    v16h a[MT];
#pragma unroll
    for (int mt = 0; mt < MT; ++mt) {
#pragma unroll
      for (int i = 0; i < 8; ++i) a[mt][i] = (_Float16)ap[mt][k0 + i];
#pragma unroll
      for (int i = 0; i < 8; ++i) a[mt][8 + i] = (_Float16)ap[mt][k0 + 8 + i];
    }

    // ---- NT B fragments, each reused by MT WMMAs ----
#pragma unroll
    for (int t = 0; t < NT; ++t) {
      v16h b = *reinterpret_cast<const v16h*>(bp[t] + k0); // 16 contiguous K
#pragma unroll
      for (int mt = 0; mt < MT; ++mt) {
        acc[mt][t] = __builtin_amdgcn_wmma_f32_16x16x32_f16(
            false, a[mt], false, b, (short)0, acc[mt][t], false, false);
      }
    }
  }

  // ---- epilogue: bias (+ReLU), store f16 and/or f32 ----
#pragma unroll
  for (int t = 0; t < NT; ++t) {
    int n = n0 + t * 16 + l15;
    float bv = bias ? bias[n] : 0.0f;
#pragma unroll
    for (int mt = 0; mt < MT; ++mt) {
#pragma unroll
      for (int r = 0; r < 8; ++r) {
        int m = m0 + mt * 16 + r + half * 8;
        float v = acc[mt][t][r] + bv;
        if (RELU) v = v > 0.0f ? v : 0.0f;
        size_t o = (size_t)m * N + n;
        if (out16) out16[o] = (_Float16)v;
        if (out32) out32[o] = v;
      }
    }
  }
}

// ---------------------------------------------------------------------------
// RVQ: per-thread row in registers; one codebook at a time in LDS.
// dist = ||c||^2 - 2 r.c  (row-constant ||r||^2 dropped for argmin)
// Deterministic loss: per-block partial sums, reduced by reduce_loss_kernel.
// ---------------------------------------------------------------------------
__global__ __launch_bounds__(256) void rvq_kernel(
    const float* __restrict__ z, const float* __restrict__ codebooks,
    float* __restrict__ zq_out, float* __restrict__ idx_out,
    _Float16* __restrict__ zq16, float* __restrict__ partial, int B) {
  __shared__ float s_cb[CB_SIZE * LAT_D];   // 32 KB
  __shared__ float s_norm[CB_SIZE];
  __shared__ float s_red[256];

  const int tid = threadIdx.x;
  const int row = blockIdx.x * 256 + tid;
  const bool valid = row < B;

  float r[LAT_D], zq[LAT_D];
#pragma unroll
  for (int d = 0; d < LAT_D; ++d) {
    r[d]  = valid ? z[(size_t)row * LAT_D + d] : 0.0f;
    zq[d] = 0.0f;
  }
  float loss = 0.0f;

  for (int c = 0; c < NUM_CB; ++c) {
    __syncthreads();
    const float* cb = codebooks + (size_t)c * CB_SIZE * LAT_D;
    for (int i = tid; i < CB_SIZE * LAT_D; i += 256) s_cb[i] = cb[i];
    __syncthreads();
    {   // 256 threads <-> 256 codes: squared norms
      float s = 0.0f;
#pragma unroll
      for (int d = 0; d < LAT_D; ++d) {
        float v = s_cb[tid * LAT_D + d];
        s = fmaf(v, v, s);
      }
      s_norm[tid] = s;
    }
    __syncthreads();

    float best = 3.4e38f;
    int   bi   = 0;
    for (int k = 0; k < CB_SIZE; ++k) {
      float dot = 0.0f;
#pragma unroll
      for (int d = 0; d < LAT_D; ++d) dot = fmaf(r[d], s_cb[k * LAT_D + d], dot);
      float dist = s_norm[k] - 2.0f * dot;
      if (dist < best) { best = dist; bi = k; }   // first-min like jnp.argmin
    }

    float ls = 0.0f;
#pragma unroll
    for (int d = 0; d < LAT_D; ++d) {
      float q  = s_cb[bi * LAT_D + d];
      float df = q - r[d];
      ls = fmaf(df, df, ls);
      zq[d] += q;
      r[d]  -= q;
    }
    if (valid) {
      loss += ls;
      idx_out[(size_t)row * NUM_CB + c] = (float)bi;
    }
  }

  if (valid) {
#pragma unroll
    for (int d = 0; d < LAT_D; ++d) {
      size_t o = (size_t)row * LAT_D + d;
      zq_out[o] = zq[d];
      zq16[o]   = (_Float16)zq[d];
    }
  }

  s_red[tid] = loss;
  __syncthreads();
  for (int s = 128; s > 0; s >>= 1) {
    if (tid < s) s_red[tid] += s_red[tid + s];
    __syncthreads();
  }
  if (tid == 0) partial[blockIdx.x] = s_red[0];
}

__global__ __launch_bounds__(256) void reduce_loss_kernel(
    const float* __restrict__ partial, int n, float* __restrict__ out,
    float scale) {
  __shared__ float s_red[256];
  const int tid = threadIdx.x;
  float a = 0.0f;
  for (int i = tid; i < n; i += 256) a += partial[i];
  s_red[tid] = a;
  __syncthreads();
  for (int s = 128; s > 0; s >>= 1) {
    if (tid < s) s_red[tid] += s_red[tid + s];
    __syncthreads();
  }
  if (tid == 0) out[0] = s_red[0] * scale;
}

// ---------------------------------------------------------------------------
// Host launcher
// ---------------------------------------------------------------------------
extern "C" void kernel_launch(void* const* d_in, const int* in_sizes, int n_in,
                              void* d_out, int out_size, void* d_ws, size_t ws_size,
                              hipStream_t stream) {
  (void)in_sizes; (void)n_in; (void)out_size; (void)ws_size;
  const int B = BATCH_N;

  // inputs: 0:x, 1-4:enc_W, 5-8:enc_b, 9-12:dec_W, 13-16:dec_b, 17:codebooks
  const float* x    = (const float*)d_in[0];
  const float* eW[4] = {(const float*)d_in[1], (const float*)d_in[2],
                        (const float*)d_in[3], (const float*)d_in[4]};
  const float* eB[4] = {(const float*)d_in[5], (const float*)d_in[6],
                        (const float*)d_in[7], (const float*)d_in[8]};
  const float* dW[4] = {(const float*)d_in[9],  (const float*)d_in[10],
                        (const float*)d_in[11], (const float*)d_in[12]};
  const float* dB[4] = {(const float*)d_in[13], (const float*)d_in[14],
                        (const float*)d_in[15], (const float*)d_in[16]};
  const float* codebooks = (const float*)d_in[17];

  // output layout: x_recon [B*512] | loss [1] | z [B*32] | z_q [B*32] | idx [B*8]
  float* out       = (float*)d_out;
  float* o_xrecon  = out;
  float* o_loss    = out + (size_t)B * 512;
  float* o_z       = o_loss + 1;
  float* o_zq      = o_z + (size_t)B * 32;
  float* o_idx     = o_zq + (size_t)B * 32;

  // workspace layout (256B-aligned regions)
  char* p = (char*)d_ws;
  auto alloc = [&](size_t bytes) -> void* {
    void* r = (void*)p;
    p += (bytes + 255) & ~(size_t)255;
    return r;
  };
  _Float16* H1  = (_Float16*)alloc((size_t)B * 512 * 2);  // enc h1 / dec h3
  _Float16* H2  = (_Float16*)alloc((size_t)B * 256 * 2);  // enc h2 / dec h2
  _Float16* H3  = (_Float16*)alloc((size_t)B * 128 * 2);  // enc h3 / dec h1
  _Float16* ZQ  = (_Float16*)alloc((size_t)B * 32 * 2);   // z_q f16 (dec input)
  const int eK[4] = {512, 512, 256, 128}, eN[4] = {512, 256, 128, 32};
  const int dK[4] = {32, 128, 256, 512},  dN[4] = {128, 256, 512, 512};
  _Float16* eWT[4]; _Float16* dWT[4];
  for (int i = 0; i < 4; ++i) eWT[i] = (_Float16*)alloc((size_t)eK[i] * eN[i] * 2);
  for (int i = 0; i < 4; ++i) dWT[i] = (_Float16*)alloc((size_t)dK[i] * dN[i] * 2);
  float* partial = (float*)alloc(512 * sizeof(float));

  // 1) weight conversion (transpose to [N,K], f32 -> f16)
  for (int i = 0; i < 4; ++i) {
    int sz = eK[i] * eN[i];
    convert_wt_kernel<<<(sz + 255) / 256, 256, 0, stream>>>(eW[i], eWT[i], eK[i], eN[i]);
  }
  for (int i = 0; i < 4; ++i) {
    int sz = dK[i] * dN[i];
    convert_wt_kernel<<<(sz + 255) / 256, 256, 0, stream>>>(dW[i], dWT[i], dK[i], dN[i]);
  }

  const int MB = B / 256;  // grid.y (row blocks of 256)

  // 2) encoder
  gemm_wmma_kernel<float, true, 4><<<dim3(512 / 64, MB), 256, 0, stream>>>(
      x, eWT[0], eB[0], B, 512, 512, H1, nullptr);
  gemm_wmma_kernel<_Float16, true, 4><<<dim3(256 / 64, MB), 256, 0, stream>>>(
      H1, eWT[1], eB[1], B, 256, 512, H2, nullptr);
  gemm_wmma_kernel<_Float16, true, 4><<<dim3(128 / 64, MB), 256, 0, stream>>>(
      H2, eWT[2], eB[2], B, 128, 256, H3, nullptr);
  gemm_wmma_kernel<_Float16, false, 2><<<dim3(1, MB), 256, 0, stream>>>(
      H3, eWT[3], eB[3], B, 32, 128, (_Float16*)nullptr, o_z);

  // 3) residual vector quantization
  rvq_kernel<<<B / 256, 256, 0, stream>>>(o_z, codebooks, o_zq, o_idx, ZQ, partial, B);
  const float scale = 0.25f / (8.0f * (float)B * 32.0f);  // COMMITMENT_COST/(n_cb*B*D)
  reduce_loss_kernel<<<1, 256, 0, stream>>>(partial, B / 256, o_loss, scale);

  // 4) decoder
  gemm_wmma_kernel<_Float16, true, 4><<<dim3(128 / 64, MB), 256, 0, stream>>>(
      ZQ, dWT[0], dB[0], B, 128, 32, H3, nullptr);
  gemm_wmma_kernel<_Float16, true, 4><<<dim3(256 / 64, MB), 256, 0, stream>>>(
      H3, dWT[1], dB[1], B, 256, 128, H2, nullptr);
  gemm_wmma_kernel<_Float16, true, 4><<<dim3(512 / 64, MB), 256, 0, stream>>>(
      H2, dWT[2], dB[2], B, 512, 256, H1, nullptr);
  gemm_wmma_kernel<_Float16, false, 4><<<dim3(512 / 64, MB), 256, 0, stream>>>(
      H1, dWT[3], dB[3], B, 512, 512, (_Float16*)nullptr, o_xrecon);
}
